// SparseAttention_62955630624779
// MI455X (gfx1250) — compile-verified, matrix operand order checked by reading, not run on hardware
//
#include <hip/hip_runtime.h>
#include <hip/hip_bf16.h>

typedef __attribute__((ext_vector_type(16))) _Float16 v16h;
typedef __attribute__((ext_vector_type(8)))  _Float16 v8h;
typedef __attribute__((ext_vector_type(8)))  float    v8f;

#define HEAD_DIM 64
#define S_LEN    512
#define N_HEADS  16
#define N_SLOTS  32      // E(8) * CAP(4)
#define BC       32      // keys per iteration
#define N_TILES  (S_LEN / BC)
#define BR       128     // query rows per block (32 per wave, 2 groups of 16)
#define MASK_PEN 1000000.0f

// ---------------- DPP16 row reductions (16-lane rows == our half-wave groups) ----
template <int CTRL>
__device__ __forceinline__ float dpp_f32(float x) {
    int r = __builtin_amdgcn_update_dpp(0, __builtin_bit_cast(int, x),
                                        CTRL, 0xF, 0xF, true);
    return __builtin_bit_cast(float, r);
}
// row_ror:1/2/4/8 = 0x121/0x122/0x124/0x128 — rotate-reduce over 16 lanes
__device__ __forceinline__ float rowmax16(float x) {
    x = fmaxf(x, dpp_f32<0x121>(x));
    x = fmaxf(x, dpp_f32<0x122>(x));
    x = fmaxf(x, dpp_f32<0x124>(x));
    x = fmaxf(x, dpp_f32<0x128>(x));
    return x;
}
__device__ __forceinline__ float rowsum16(float x) {
    x += dpp_f32<0x121>(x);
    x += dpp_f32<0x122>(x);
    x += dpp_f32<0x124>(x);
    x += dpp_f32<0x128>(x);
    return x;
}

// ---------------- async global -> LDS (CDNA5, ASYNCcnt-tracked DMA) ----------
__device__ __forceinline__ void async_load_b128(unsigned lds_off, const float* gaddr) {
    asm volatile("global_load_async_to_lds_b128 %0, %1, off"
                 :: "v"(lds_off), "v"((unsigned long long)(uintptr_t)gaddr)
                 : "memory");
}
__device__ __forceinline__ void wait_async0(void) {
    asm volatile("s_wait_asynccnt 0x0" ::: "memory");
}

// ---------------- fragment builders ------------------------------------------
// A/B frag from f32 row: halves 0..7 = p[0..7], halves 8..15 = p[16..23]
__device__ __forceinline__ v16h frag_from_f32(const float* p) {
    float4 a0 = *(const float4*)(p);
    float4 a1 = *(const float4*)(p + 4);
    float4 b0 = *(const float4*)(p + 16);
    float4 b1 = *(const float4*)(p + 20);
    v16h r;
    r[0]=(_Float16)a0.x;  r[1]=(_Float16)a0.y;  r[2]=(_Float16)a0.z;  r[3]=(_Float16)a0.w;
    r[4]=(_Float16)a1.x;  r[5]=(_Float16)a1.y;  r[6]=(_Float16)a1.z;  r[7]=(_Float16)a1.w;
    r[8]=(_Float16)b0.x;  r[9]=(_Float16)b0.y;  r[10]=(_Float16)b0.z; r[11]=(_Float16)b0.w;
    r[12]=(_Float16)b1.x; r[13]=(_Float16)b1.y; r[14]=(_Float16)b1.z; r[15]=(_Float16)b1.w;
    return r;
}
// V B-frag: column n of B = output dim d; contraction k = key (row-major f32 tile)
__device__ __forceinline__ v16h vfrag(const float (*Vf)[HEAD_DIM], int d, int hi) {
    v16h r;
    #pragma unroll
    for (int j = 0; j < 8; ++j) {
        r[j]     = (_Float16)Vf[hi * 8 + j][d];
        r[8 + j] = (_Float16)Vf[16 + hi * 8 + j][d];
    }
    return r;
}
// P A-frag from per-wave f16 scratch
__device__ __forceinline__ v16h lds_frag16(const _Float16* lo, const _Float16* hi) {
    v8h a = *(const v8h*)lo;
    v8h b = *(const v8h*)hi;
    return __builtin_shufflevector(a, b, 0,1,2,3,4,5,6,7,8,9,10,11,12,13,14,15);
}

#define WMMA_F16(A, B, C) \
    __builtin_amdgcn_wmma_f32_16x16x32_f16(false, (A), false, (B), (short)0, (C), false, false)

__global__ __launch_bounds__(128)
void sparse_attn_fa_kernel(const float* __restrict__ Q,
                           const float* __restrict__ K,
                           const float* __restrict__ V,
                           const int*   __restrict__ idx_list,
                           const float* __restrict__ mask,
                           float*       __restrict__ out) {
    __shared__ __align__(16) float Kbuf[2][BC][HEAD_DIM];   // 2 x 8 KB, f32
    __shared__ __align__(16) float Vbuf[2][BC][HEAD_DIM];   // 2 x 8 KB, f32
    __shared__ __align__(16) _Float16 Pw[4][2][16][BC];     // per-wave/group P scratch, 8 KB

    const int tid  = threadIdx.x;
    const int lane = tid & 31;
    const int w    = tid >> 5;       // wave id 0..3
    const int n    = lane & 15;      // column within a 16-wide tile
    const int hi   = lane >> 4;      // lane half

    const int bid  = blockIdx.x;
    const int qt   = bid & 3;               // q-tile (128 rows / block)
    const int h    = (bid >> 2) & 15;       // head
    const int slot = bid >> 6;              // expert*CAP + cap (0..31)

    const int b = idx_list[slot];           // routed batch index
    const size_t bh = (size_t)(b * N_HEADS + h);
    const float* Qbh = Q + bh * (size_t)S_LEN * HEAD_DIM;
    const float* Kbh = K + bh * (size_t)S_LEN * HEAD_DIM;
    const float* Vbh = V + bh * (size_t)S_LEN * HEAD_DIM;
    const float* mb  = mask + (size_t)b * S_LEN;

    const int q0 = qt * BR + w * 32;        // this wave's first query row (32 rows)

    // ---- issue async DMA for tile 0 (each thread covers 4x16B of K and of V) ----
    {
        #pragma unroll
        for (int j = 0; j < 4; ++j) {
            const int s   = tid + 128 * j;
            const int key = s >> 4;
            const int d   = (s & 15) * 4;
            const size_t g = (size_t)key * HEAD_DIM + d;
            async_load_b128((unsigned)(uintptr_t)&Kbuf[0][key][d], Kbh + g);
            async_load_b128((unsigned)(uintptr_t)&Vbuf[0][key][d], Vbh + g);
        }
    }

    // ---- Q A-fragments: 2 row-groups x 2 d-chunks, held for the whole kernel ----
    v16h aq[2][2];
    #pragma unroll
    for (int g = 0; g < 2; ++g) {
        const float* qrow = Qbh + (size_t)(q0 + g * 16 + n) * HEAD_DIM;
        aq[g][0] = frag_from_f32(qrow + hi * 8);          // d = 0..31
        aq[g][1] = frag_from_f32(qrow + 32 + hi * 8);     // d = 32..63
    }

    v8f o[2][4] = {};
    float mrow[2][8], lrow[2][8];
    #pragma unroll
    for (int g = 0; g < 2; ++g)
        #pragma unroll
        for (int r = 0; r < 8; ++r) { mrow[g][r] = -1e30f; lrow[g][r] = 0.0f; }

    const float scale = 0.125f; // 1/sqrt(64)

    for (int kt = 0; kt < N_TILES; ++kt) {
        const int buf = kt & 1;

        wait_async0();      // this wave's async writes for tile kt are in LDS
        __syncthreads();    // all waves' writes visible; prior compute done

        // prefetch tile kt+1 into the other buffer (its readers are all past the barrier)
        if (kt + 1 < N_TILES) {
            const int kb2 = (kt + 1) * BC;
            #pragma unroll
            for (int j = 0; j < 4; ++j) {
                const int s   = tid + 128 * j;
                const int key = s >> 4;
                const int d   = (s & 15) * 4;
                const size_t g = (size_t)(kb2 + key) * HEAD_DIM + d;
                async_load_b128((unsigned)(uintptr_t)&Kbuf[buf ^ 1][key][d], Kbh + g);
                async_load_b128((unsigned)(uintptr_t)&Vbuf[buf ^ 1][key][d], Vbh + g);
            }
        }

        const int kbase = kt * BC;

        // ---- shared B-fragments: built ONCE per wave, reused by both row groups ----
        v16h b00 = frag_from_f32(&Kbuf[buf][n][hi * 8]);
        v16h b01 = frag_from_f32(&Kbuf[buf][n][32 + hi * 8]);
        v16h b10 = frag_from_f32(&Kbuf[buf][16 + n][hi * 8]);
        v16h b11 = frag_from_f32(&Kbuf[buf][16 + n][32 + hi * 8]);
        v16h bv0 = vfrag(Vbuf[buf], n,      hi);
        v16h bv1 = vfrag(Vbuf[buf], 16 + n, hi);
        v16h bv2 = vfrag(Vbuf[buf], 32 + n, hi);
        v16h bv3 = vfrag(Vbuf[buf], 48 + n, hi);

        const float pen0 = MASK_PEN * (1.0f - mb[kbase + n]);
        const float pen1 = MASK_PEN * (1.0f - mb[kbase + 16 + n]);

        #pragma unroll
        for (int g = 0; g < 2; ++g) {
            // ---- scores: S = Q * K^T (two 16x16 tiles over 32 keys) ----
            v8f s0 = {}, s1 = {};
            s0 = WMMA_F16(aq[g][0], b00, s0);
            s0 = WMMA_F16(aq[g][1], b01, s0);
            s1 = WMMA_F16(aq[g][0], b10, s1);
            s1 = WMMA_F16(aq[g][1], b11, s1);

            // ---- online softmax (DPP16 row reductions) ----
            #pragma unroll
            for (int r = 0; r < 8; ++r) {
                float v0 = s0[r] * scale - pen0;
                float v1 = s1[r] * scale - pen1;
                float mx = rowmax16(fmaxf(v0, v1));
                const float mnew = fmaxf(mrow[g][r], mx);
                const float p0 = __expf(v0 - mnew);
                const float p1 = __expf(v1 - mnew);
                const float sf = __expf(mrow[g][r] - mnew);
                const float rs = rowsum16(p0 + p1);
                lrow[g][r] = lrow[g][r] * sf + rs;
                mrow[g][r] = mnew;
                o[g][0][r] *= sf; o[g][1][r] *= sf;
                o[g][2][r] *= sf; o[g][3][r] *= sf;
                const int mr = r + hi * 8;
                Pw[w][g][mr][n]      = (_Float16)p0;
                Pw[w][g][mr][16 + n] = (_Float16)p1;
            }

            // ---- O += P * V (contraction over 32 keys, 4 output d-tiles) ----
            // Same-wave LDS ops are in-order: Pw stores above are visible here.
            v16h pa = lds_frag16(&Pw[w][g][n][hi * 8], &Pw[w][g][n][16 + hi * 8]);
            o[g][0] = WMMA_F16(pa, bv0, o[g][0]);
            o[g][1] = WMMA_F16(pa, bv1, o[g][1]);
            o[g][2] = WMMA_F16(pa, bv2, o[g][2]);
            o[g][3] = WMMA_F16(pa, bv3, o[g][3]);
        }
    }

    // ---- epilogue: divide by row sum, store fp32 ----
    #pragma unroll
    for (int g = 0; g < 2; ++g) {
        float* obase = out + ((size_t)(slot * N_HEADS + h) * S_LEN + q0 + g * 16) * HEAD_DIM;
        #pragma unroll
        for (int r = 0; r < 8; ++r) {
            const int mr = r + hi * 8;
            const float inv = 1.0f / lrow[g][r];
            float* orow = obase + (size_t)mr * HEAD_DIM;
            orow[n]      = o[g][0][r] * inv;
            orow[16 + n] = o[g][1][r] * inv;
            orow[32 + n] = o[g][2][r] * inv;
            orow[48 + n] = o[g][3][r] * inv;
        }
    }
}

extern "C" void kernel_launch(void* const* d_in, const int* in_sizes, int n_in,
                              void* d_out, int out_size, void* d_ws, size_t ws_size,
                              hipStream_t stream) {
    const float* Q        = (const float*)d_in[0];
    const float* K        = (const float*)d_in[1];
    const float* V        = (const float*)d_in[2];
    const int*   idx_list = (const int*)d_in[3];
    const float* mask     = (const float*)d_in[4];
    float* out = (float*)d_out;

    dim3 grid(N_SLOTS * N_HEADS * (S_LEN / BR));   // 2048 blocks
    dim3 block(128);                               // 4 waves (wave32), 32 q-rows each
    sparse_attn_fa_kernel<<<grid, block, 0, stream>>>(Q, K, V, idx_list, mask, out);
}
